// BoundaryRepresentation_19370302505482
// MI455X (gfx1250) — matrix operand order repaired and checked by reading, not compiled, not run on hardware
//
#include <hip/hip_runtime.h>
#include <hip/hip_bf16.h>

// ---------------------------------------------------------------------------
// Types for CDNA5 WMMA (wave32): v_wmma_f32_16x16x32_bf16
// ---------------------------------------------------------------------------
typedef __attribute__((ext_vector_type(16))) __bf16 v16bf;
typedef __attribute__((ext_vector_type(8)))  __bf16 v8bf;
typedef __attribute__((ext_vector_type(8)))  float  v8f;

static __device__ inline v8f vzero8() {
  v8f z;
#pragma unroll
  for (int i = 0; i < 8; ++i) z[i] = 0.f;
  return z;
}

union frag_u { v16bf v; v8bf h[2]; };

// A-matrix 16x32 bf16 fragment (ISA 7.12.2): lanes 0-15 hold M=lane,
// K={0..7,16..23}; lanes 16-31 hold M=lane-16, K={8..15,24..31}.
static __device__ inline v16bf load_a_frag(const __bf16* base, int stride,
                                           int row0, int half, int l16, int k0) {
  const __bf16* p = base + (row0 + l16) * stride + k0 + half * 8;
  frag_u u;
  u.h[0] = *reinterpret_cast<const v8bf*>(p);
  u.h[1] = *reinterpret_cast<const v8bf*>(p + 16);
  return u.v;
}

// B-matrix 32x16 bf16 fragment: lanes 0-15 = col, K={0..7,16..23}; lanes 16-31 = col,
// K={8..15,24..31}. LDS weight tile is col-major [n][stride-over-k] bf16.
static __device__ inline v16bf load_b_frag(const __bf16* base, int stride,
                                           int col0, int half, int l16, int k0) {
  const __bf16* p = base + (col0 + l16) * stride + k0 + half * 8;
  frag_u u;
  u.h[0] = *reinterpret_cast<const v8bf*>(p);
  u.h[1] = *reinterpret_cast<const v8bf*>(p + 16);
  return u.v;
}

static __device__ inline v8f wmma_bf16(v16bf a, v16bf b, v8f c) {
  return __builtin_amdgcn_wmma_f32_16x16x32_bf16(false, a, false, b,
                                                 (short)0, c, false, false);
}

// ---------------------------------------------------------------------------
// CDNA5 async memory->LDS copy (ASYNCcnt-tracked, GV addressing, saddr=off).
// lds_off: LDS byte address = low 32 bits of flat shared-aperture address.
// ---------------------------------------------------------------------------
static __device__ inline void async_copy_b128(unsigned lds_off,
                                              unsigned long long gaddr) {
  asm volatile("global_load_async_to_lds_b128 %0, %1, off"
               :: "v"(lds_off), "v"(gaddr)
               : "memory");
}
static __device__ inline void wait_async0() {
  asm volatile("s_wait_asynccnt 0x0" ::: "memory");
}
static __device__ inline unsigned lds_offset_of(const void* p) {
  return (unsigned)(unsigned long long)p;   // addr[31:0] = LDS offset
}

// ---------------------------------------------------------------------------
// Weight pre-tiling: W[K,N] f32 -> bf16 LDS-image tiles, tile = 128 n x 32 k,
// col-major [n][40] (padded). Tile order: nb-major, then ks.
// out[((nb*(K/32)+ks)*128 + n)*40 + k]
// ---------------------------------------------------------------------------
__global__ __launch_bounds__(256) void tile_weights_kernel(
    const float* __restrict__ W, int N, int K, __bf16* __restrict__ out) {
  int idx = blockIdx.x * 256 + threadIdx.x;
  if (idx >= N * K) return;
  int n_g = idx % N, k_g = idx / N;     // consecutive threads -> consec n (coalesced)
  int nb = n_g >> 7, n = n_g & 127;
  int ks = k_g >> 5, k = k_g & 31;
  out[(((nb * (K >> 5)) + ks) * 128 + n) * 40 + k] = (__bf16)W[(size_t)k_g * N + n_g];
}

// Pair-kernel W1 [256,256] f32 -> 4 K-slices of [256 n][72 k-pad] bf16.
__global__ __launch_bounds__(256) void tile_w1_kernel(
    const float* __restrict__ w1, __bf16* __restrict__ out) {
  int idx = blockIdx.x * 256 + threadIdx.x;   // 65536 total
  int n = idx & 255, k_g = idx >> 8;
  int s4 = k_g >> 6, k = k_g & 63;
  out[(s4 * 256 + n) * 72 + k] = (__bf16)w1[k_g * 256 + n];
}

// ---------------------------------------------------------------------------
// Generic GEMM: C[M,N] = act(A[M,K] @ W[K,N] + bias)
// A f32 global (converted to bf16 while staging); W pre-tiled bf16, staged via
// GLOBAL_LOAD_ASYNC_TO_LDS_B128. Block tile 64x128, 8 waves as 2(M)x4(N).
// Requires M%64==0, N%128==0, K%32==0.
// ---------------------------------------------------------------------------
__global__ __launch_bounds__(256) void gemm_bf16_kernel(
    const float* __restrict__ A, int lda,
    const __bf16* __restrict__ Wt,
    const float* __restrict__ bias,
    float* __restrict__ C, int ldc,
    int K, int relu) {
  const int tid  = threadIdx.x;
  const int lane = tid & 31;
  const int wv   = tid >> 5;
  const int half = lane >> 4;
  const int l16  = lane & 15;
  const int row0 = blockIdx.x * 64;
  const int col0 = blockIdx.y * 128;
  const int mbase = (wv >> 2) * 32;  // 0 | 32
  const int nbase = (wv & 3) * 32;   // 0,32,64,96

  __shared__ __align__(16) __bf16 sA[64 * 40];
  __shared__ __align__(16) __bf16 sW[128 * 40];
  const unsigned ldsW = lds_offset_of(&sW[0]);

  v8f acc[2][2];
#pragma unroll
  for (int a = 0; a < 2; ++a)
#pragma unroll
    for (int b = 0; b < 2; ++b) acc[a][b] = vzero8();

  for (int kk = 0; kk < K; kk += 32) {
    __syncthreads();
    // Async-stage pre-tiled W slice: 128*40 bf16 = 10240 B contiguous
    {
      unsigned long long src = (unsigned long long)
          (Wt + (size_t)(blockIdx.y * (K >> 5) + (kk >> 5)) * (128 * 40));
      unsigned boff = (unsigned)tid * 16u;
      async_copy_b128(ldsW + boff,        src + boff);
      async_copy_b128(ldsW + boff + 4096, src + boff + 4096);
      if (tid < 128) async_copy_b128(ldsW + boff + 8192, src + boff + 8192);
    }
    // Prefetch next A slice into cache while staging this one
    if (kk + 32 < K)
      __builtin_prefetch(A + (size_t)(row0 + (tid & 63)) * lda + kk + 32, 0, 0);
    // Stage A tile 64x32 (f32 -> bf16)
#pragma unroll
    for (int it = 0; it < 8; ++it) {
      int idx = tid + it * 256;
      int r = idx >> 5, k = idx & 31;
      sA[r * 40 + k] = (__bf16)A[(size_t)(row0 + r) * lda + kk + k];
    }
    wait_async0();
    __syncthreads();

    v16bf af[2];
#pragma unroll
    for (int mt = 0; mt < 2; ++mt)
      af[mt] = load_a_frag(sA, 40, mbase + mt * 16, half, l16, 0);
#pragma unroll
    for (int nt = 0; nt < 2; ++nt) {
      v16bf bf_ = load_b_frag(sW, 40, nbase + nt * 16, half, l16, 0);
#pragma unroll
      for (int mt = 0; mt < 2; ++mt)
        acc[mt][nt] = wmma_bf16(af[mt], bf_, acc[mt][nt]);
    }
  }

  // Epilogue. C/D layout: VGPR v -> row m = v + 8*half, col = l16.
#pragma unroll
  for (int mt = 0; mt < 2; ++mt)
#pragma unroll
    for (int nt = 0; nt < 2; ++nt) {
      int coln = col0 + nbase + nt * 16 + l16;
      float bv = bias ? bias[coln] : 0.f;
#pragma unroll
      for (int v = 0; v < 8; ++v) {
        int rowm = row0 + mbase + mt * 16 + v + half * 8;
        float val = acc[mt][nt][v] + bv;
        if (relu) val = fmaxf(val, 0.f);
        C[(size_t)rowm * ldc + coln] = val;
      }
    }
}

// ---------------------------------------------------------------------------
// Fused pair scoring: block = (b, i, j0..j0+63):
//   x[r] = Bs[b,i] + Be[b,j0+r] + pos[|i-(j0+r)|]  -> LayerNorm -> bf16 LDS
//   h    = relu(a @ W1 + b1)      (64x256x256 GEMM, WMMA bf16)
//   S[r] = h . w2 + b2            (shfl butterfly + LDS f32 atomics)
//   G[b,i,j] = S - 10000*(1 - mask2)
// W1 pre-tiled bf16; K-slices staged with async LDS copies.
// ---------------------------------------------------------------------------
__global__ __launch_bounds__(256) void pair_score_kernel(
    const float* __restrict__ Bs, const float* __restrict__ Be,
    const float* __restrict__ pos,
    const float* __restrict__ ln_g, const float* __restrict__ ln_b,
    const __bf16* __restrict__ w1t, const float* __restrict__ b1,
    const float* __restrict__ w2, const float* __restrict__ b2,
    const float* __restrict__ masks, float* __restrict__ G) {
  const int L = 256;
  const int tid = threadIdx.x;
  int blk = blockIdx.x;
  int j0 = (blk & 3) * 64;
  int tmp = blk >> 2;
  int i = tmp & 255;
  int b = tmp >> 8;

  __shared__ __align__(16) __bf16 sA[64 * 264];  // LN'd rows, full K=256
  __shared__ __align__(16) __bf16 sW[256 * 72];  // W1 K-slice (64 k), col-major
  __shared__ float sS[64];
  __shared__ float sRed[64 * 8];
  const unsigned ldsW = lds_offset_of(&sW[0]);

  // ---- Phase 1: build LN'd activation rows (4 threads per row) ----
  {
    int r = tid >> 2, q = tid & 3;
    const float* bs_row  = Bs + (b * L + i) * 256;
    int j = j0 + r;
    const float* be_row  = Be + (b * L + j) * 256;
    int d = (i > j) ? (i - j) : (j - i);
    const float* pos_row = pos + d * 256;
    float s = 0.f, s2 = 0.f;
    for (int k = q * 64; k < q * 64 + 64; ++k) {
      float x = bs_row[k] + be_row[k] + pos_row[k];
      s += x; s2 += x * x;
    }
    sRed[r * 8 + q] = s;
    sRed[r * 8 + 4 + q] = s2;
    if (tid < 64) sS[tid] = 0.f;
    __syncthreads();
    float mean = (sRed[r*8+0] + sRed[r*8+1] + sRed[r*8+2] + sRed[r*8+3]) * (1.f/256.f);
    float ms   = (sRed[r*8+4] + sRed[r*8+5] + sRed[r*8+6] + sRed[r*8+7]) * (1.f/256.f);
    float rstd = rsqrtf(fmaxf(ms - mean * mean, 0.f) + 1e-12f);
    for (int k = q * 64; k < q * 64 + 64; ++k) {
      float x = bs_row[k] + be_row[k] + pos_row[k];
      float y = (x - mean) * rstd * ln_g[k] + ln_b[k];
      sA[r * 264 + k] = (__bf16)y;
    }
  }

  // ---- Phase 2: 64x256 GEMM vs W1 (K=256), WMMA bf16 ----
  const int lane = tid & 31;
  const int wv = tid >> 5;
  const int half = lane >> 4;
  const int l16 = lane & 15;
  const int mbase = (wv >> 2) * 32;  // 0 | 32
  const int nbase = (wv & 3) * 64;   // 0,64,128,192

  v8f acc[2][4];
#pragma unroll
  for (int a = 0; a < 2; ++a)
#pragma unroll
    for (int n = 0; n < 4; ++n) acc[a][n] = vzero8();

  for (int s4 = 0; s4 < 4; ++s4) {
    __syncthreads();
    // Async-stage W1 slice: 256*72 bf16 = 36864 B = 9 x (256 lanes x 16 B)
    {
      unsigned long long src = (unsigned long long)(w1t + s4 * (256 * 72));
#pragma unroll
      for (int it = 0; it < 9; ++it) {
        unsigned boff = (unsigned)(tid + it * 256) * 16u;
        async_copy_b128(ldsW + boff, src + boff);
      }
    }
    wait_async0();
    __syncthreads();
    int ks0 = s4 * 64;
#pragma unroll
    for (int kk = 0; kk < 64; kk += 32) {
      v16bf af[2];
      af[0] = load_a_frag(sA, 264, mbase, half, l16, ks0 + kk);
      af[1] = load_a_frag(sA, 264, mbase + 16, half, l16, ks0 + kk);
#pragma unroll
      for (int nt = 0; nt < 4; ++nt) {
        v16bf bf_ = load_b_frag(sW, 72, nbase + nt * 16, half, l16, kk);
        acc[0][nt] = wmma_bf16(af[0], bf_, acc[0][nt]);
        acc[1][nt] = wmma_bf16(af[1], bf_, acc[1][nt]);
      }
    }
  }
  __syncthreads();

  // ---- Phase 3: relu + dot with w2, butterfly-reduce, LDS atomic add ----
  float b1v[4], w2v[4];
#pragma unroll
  for (int nt = 0; nt < 4; ++nt) {
    int col = nbase + nt * 16 + l16;
    b1v[nt] = b1[col];
    w2v[nt] = w2[col];
  }
#pragma unroll
  for (int mt = 0; mt < 2; ++mt)
#pragma unroll
    for (int v = 0; v < 8; ++v) {
      float p = 0.f;
#pragma unroll
      for (int nt = 0; nt < 4; ++nt) {
        float h = fmaxf(acc[mt][nt][v] + b1v[nt], 0.f);
        p += h * w2v[nt];
      }
#pragma unroll
      for (int off = 1; off < 16; off <<= 1) p += __shfl_xor(p, off, 32);
      if (l16 == 0)
        atomicAdd(&sS[mbase + mt * 16 + v + half * 8], p);
    }
  __syncthreads();

  // ---- Phase 4: mask + write G ----
  if (tid < 64) {
    int jj = j0 + tid;
    float S = sS[tid] + b2[0];
    float tri = (i <= jj && (jj - i) <= 30) ? 1.f : 0.f;
    float m2 = masks[b * L + i] * masks[b * L + jj] * tri;
    G[(b * L + i) * L + jj] = S - 10000.f * (1.f - m2);
  }
}

// ---------------------------------------------------------------------------
// Softmax over 256 elements. transpose=0: rows of G; transpose=1: columns
// (softmax over i of G[b,i,j]); output always row-major P[b,r,t].
// ---------------------------------------------------------------------------
__global__ __launch_bounds__(256) void softmax256_kernel(
    const float* __restrict__ G, float* __restrict__ P, int transpose) {
  const int L = 256;
  int br = blockIdx.x;
  int b = br >> 8, r = br & 255;
  int t = threadIdx.x;
  float x = transpose ? G[(b * L + t) * L + r] : G[(b * L + r) * L + t];
  __shared__ float red[256];
  red[t] = x;
  __syncthreads();
  for (int s = 128; s > 0; s >>= 1) {
    if (t < s) red[t] = fmaxf(red[t], red[t + s]);
    __syncthreads();
  }
  float mx = red[0];
  __syncthreads();
  float e = __expf(x - mx);
  red[t] = e;
  __syncthreads();
  for (int s = 128; s > 0; s >>= 1) {
    if (t < s) red[t] += red[t + s];
    __syncthreads();
  }
  P[(b * L + r) * L + t] = e / red[0];
}

// ---------------------------------------------------------------------------
// qs = relu(H_cls @ W1 + b1) @ w2 + b2   (per batch row; tiny)
// ---------------------------------------------------------------------------
__global__ __launch_bounds__(256) void cls_score_kernel(
    const float* __restrict__ Hcls,
    const float* __restrict__ w1, const float* __restrict__ b1,
    const float* __restrict__ w2, const float* __restrict__ b2,
    float* __restrict__ out) {
  int b = blockIdx.x;
  int t = threadIdx.x;
  const float* x = Hcls + b * 768;
  float p = 0.f;
  for (int u = t; u < 768; u += 256) {
    float h = b1[u];
    for (int k = 0; k < 768; ++k) h += x[k] * w1[k * 768 + u];
    p += fmaxf(h, 0.f) * w2[u];
  }
  __shared__ float red[256];
  red[t] = p;
  __syncthreads();
  for (int s = 128; s > 0; s >>= 1) {
    if (t < s) red[t] += red[t + s];
    __syncthreads();
  }
  if (t == 0) out[b] = red[0] + b2[0];
}

// ---------------------------------------------------------------------------
extern "C" void kernel_launch(void* const* d_in, const int* in_sizes, int n_in,
                              void* d_out, int out_size, void* d_ws, size_t ws_size,
                              hipStream_t stream) {
  (void)in_sizes; (void)n_in; (void)out_size; (void)ws_size;
  const float* H_c        = (const float*)d_in[0];
  const float* H_cls      = (const float*)d_in[1];
  const float* masks      = (const float*)d_in[2];
  const float* bs_l1_w    = (const float*)d_in[3];
  const float* bs_l1_b    = (const float*)d_in[4];
  const float* bs_l2_w    = (const float*)d_in[5];
  const float* bs_l2_b    = (const float*)d_in[6];
  const float* be_l1_w    = (const float*)d_in[7];
  const float* be_l1_b    = (const float*)d_in[8];
  const float* be_l2_w    = (const float*)d_in[9];
  const float* be_l2_b    = (const float*)d_in[10];
  const float* ses_smap_w = (const float*)d_in[11];
  const float* ses_smap_b = (const float*)d_in[12];
  const float* ses_pos    = (const float*)d_in[13];
  const float* ses_ln_g   = (const float*)d_in[14];
  const float* ses_ln_b   = (const float*)d_in[15];
  const float* see_smap_w = (const float*)d_in[16];
  const float* see_smap_b = (const float*)d_in[17];
  const float* see_pos    = (const float*)d_in[18];
  const float* see_ln_g   = (const float*)d_in[19];
  const float* see_ln_b   = (const float*)d_in[20];
  const float* sss_sc_l1_w  = (const float*)d_in[21];
  const float* sss_sc_l1_b  = (const float*)d_in[22];
  const float* sss_sc_l2_w  = (const float*)d_in[23];
  const float* sss_sc_l2_b  = (const float*)d_in[24];
  const float* sss_cls_l1_w = (const float*)d_in[25];
  const float* sss_cls_l1_b = (const float*)d_in[26];
  const float* sss_cls_l2_w = (const float*)d_in[27];
  const float* sss_cls_l2_b = (const float*)d_in[28];
  const float* sse_sc_l1_w  = (const float*)d_in[29];
  const float* sse_sc_l1_b  = (const float*)d_in[30];
  const float* sse_sc_l2_w  = (const float*)d_in[31];
  const float* sse_sc_l2_b  = (const float*)d_in[32];
  const float* sse_cls_l1_w = (const float*)d_in[33];
  const float* sse_cls_l1_b = (const float*)d_in[34];
  const float* sse_cls_l2_w = (const float*)d_in[35];
  const float* sse_cls_l2_b = (const float*)d_in[36];
  const float* w2s_w = (const float*)d_in[37];
  const float* w2s_b = (const float*)d_in[38];
  const float* w2e_w = (const float*)d_in[39];
  const float* w2e_b = (const float*)d_in[40];

  // ---- Workspace layout ----
  // f32 region (2,752,512 floats)
  float* ws  = (float*)d_ws;
  float* X   = ws;              // 512*768
  float* Bs0 = ws + 393216;
  float* Be0 = ws + 786432;
  float* Bss = ws + 1179648;    // 512*256
  float* Bes = ws + 1310720;
  float* Bse = ws + 1441792;
  float* Bee = ws + 1572864;
  float* Vs  = ws + 1703936;    // 512*768
  float* Ve  = ws + 2097152;
  float* Ps  = ws + 2490368;    // 2*256*256
  float* Pe  = ws + 2621440;
  // bf16 pre-tiled weight region
  __bf16* bws     = (__bf16*)(ws + 2752512);
  __bf16* t_bs1   = bws;                 // 768x768 tiled: 6*24*5120 = 737280
  __bf16* t_bs2   = t_bs1 + 737280;
  __bf16* t_be1   = t_bs2 + 737280;
  __bf16* t_be2   = t_be1 + 737280;
  __bf16* t_w2s   = t_be2 + 737280;
  __bf16* t_w2e   = t_w2s + 737280;
  __bf16* t_smaps = t_w2e + 737280;      // K=768,N=256: 2*24*5120 = 245760
  __bf16* t_smape = t_smaps + 245760;
  __bf16* t_w1s   = t_smape + 245760;    // 4*256*72 = 73728
  __bf16* t_w1e   = t_w1s + 73728;
  __bf16* t_Vs    = t_w1e + 73728;       // per batch K=256,N=768: 6*8*5120 = 245760
  __bf16* t_Ve    = t_Vs + 2 * 245760;

  // ---- Output layout (floats): B_s, B_e, G_s, G_e, qs_s, qs_e ----
  float* out = (float*)d_out;
  float* oBs = out;
  float* oBe = out + 393216;
  float* oGs = out + 786432;
  float* oGe = out + 917504;
  float* oQs = out + 1048576;
  float* oQe = out + 1048578;

  dim3 blk(256);
  auto tileW = [&](const float* W, int N, int K, __bf16* dst) {
    tile_weights_kernel<<<dim3((N * K + 255) / 256), blk, 0, stream>>>(W, N, K, dst);
  };
  auto gemm = [&](const float* A, int lda, const __bf16* Wt,
                  const float* bias, float* C, int ldc,
                  int M, int N, int K, int relu) {
    dim3 grid(M / 64, N / 128);
    gemm_bf16_kernel<<<grid, blk, 0, stream>>>(A, lda, Wt, bias, C, ldc, K, relu);
  };

  // ---- One-time weight pre-tiling (f32 -> bf16 LDS-image layout) ----
  tileW(bs_l1_w, 768, 768, t_bs1);
  tileW(bs_l2_w, 768, 768, t_bs2);
  tileW(be_l1_w, 768, 768, t_be1);
  tileW(be_l2_w, 768, 768, t_be2);
  tileW(w2s_w,   768, 768, t_w2s);
  tileW(w2e_w,   768, 768, t_w2e);
  tileW(ses_smap_w, 256, 768, t_smaps);
  tileW(see_smap_w, 256, 768, t_smape);
  tile_w1_kernel<<<256, blk, 0, stream>>>(sss_sc_l1_w, t_w1s);
  tile_w1_kernel<<<256, blk, 0, stream>>>(sse_sc_l1_w, t_w1e);

  // ---- Boundary MLPs: B_s0, B_e0 ----
  gemm(H_c, 768, t_bs1, bs_l1_b, X,   768, 512, 768, 768, 1);
  gemm(X,   768, t_bs2, bs_l2_b, Bs0, 768, 512, 768, 768, 0);
  gemm(H_c, 768, t_be1, be_l1_b, X,   768, 512, 768, 768, 1);
  gemm(X,   768, t_be2, be_l2_b, Be0, 768, 512, 768, 768, 0);
  // ---- Span mappings (s_mapping applied to BOTH, bug-faithful) ----
  gemm(Bs0, 768, t_smaps, ses_smap_b, Bss, 256, 512, 256, 768, 0);
  gemm(Be0, 768, t_smaps, ses_smap_b, Bes, 256, 512, 256, 768, 0);
  gemm(Bs0, 768, t_smape, see_smap_b, Bse, 256, 512, 256, 768, 0);
  gemm(Be0, 768, t_smape, see_smap_b, Bee, 256, 512, 256, 768, 0);
  // ---- Value projections (both from B_s0, bug-faithful) ----
  gemm(Bs0, 768, t_w2s, w2s_b, Vs, 768, 512, 768, 768, 0);
  gemm(Bs0, 768, t_w2e, w2e_b, Ve, 768, 512, 768, 768, 0);
  // ---- Pair scoring -> G_s, G_e (dominant cost; WMMA + async staging) ----
  pair_score_kernel<<<2048, blk, 0, stream>>>(Bss, Bes, ses_pos, ses_ln_g, ses_ln_b,
      t_w1s, sss_sc_l1_b, sss_sc_l2_w, sss_sc_l2_b, masks, oGs);
  pair_score_kernel<<<2048, blk, 0, stream>>>(Bse, Bee, see_pos, see_ln_g, see_ln_b,
      t_w1e, sse_sc_l1_b, sse_sc_l2_w, sse_sc_l2_b, masks, oGe);
  // ---- Softmax (rows for G_s; columns of G_e = softmax over i of G_e^T) ----
  softmax256_kernel<<<512, blk, 0, stream>>>(oGs, Ps, 0);
  softmax256_kernel<<<512, blk, 0, stream>>>(oGe, Pe, 1);
  // ---- Re-tile V_s, V_e per batch (runtime weights for P@V) ----
  for (int b = 0; b < 2; ++b) {
    tileW(Vs + b * 196608, 768, 256, t_Vs + b * 245760);
    tileW(Ve + b * 196608, 768, 256, t_Ve + b * 245760);
  }
  // ---- B_s = P_s @ V_s ; B_e = P_e @ V_e  (per batch) ----
  for (int b = 0; b < 2; ++b) {
    gemm(Ps + b * 65536, 256, t_Vs + b * 245760, nullptr,
         oBs + b * 196608, 768, 256, 768, 256, 0);
    gemm(Pe + b * 65536, 256, t_Ve + b * 245760, nullptr,
         oBe + b * 196608, 768, 256, 768, 256, 0);
  }
  // ---- qs_s, qs_e ----
  cls_score_kernel<<<2, blk, 0, stream>>>(H_cls, sss_cls_l1_w, sss_cls_l1_b,
                                          sss_cls_l2_w, sss_cls_l2_b, oQs);
  cls_score_kernel<<<2, blk, 0, stream>>>(H_cls, sse_cls_l1_w, sse_cls_l1_b,
                                          sse_cls_l2_w, sse_cls_l2_b, oQe);
}
